// Attention_5909875000263
// MI455X (gfx1250) — matrix operand order repaired and checked by reading, not compiled, not run on hardware
//
#include <hip/hip_runtime.h>

// ---------------------------------------------------------------------------
// MI455X (gfx1250) attention layer: bf16 WMMA, double-buffered async-to-LDS
// GEMM staging, dual-descriptor TDM attention staging.
//   B=4, S=2048, H=16, Dh=64, Dm=1024
// ---------------------------------------------------------------------------

typedef __attribute__((ext_vector_type(16))) __bf16        v16bf;
typedef __attribute__((ext_vector_type(8)))  float         v8f;
typedef __attribute__((ext_vector_type(4)))  unsigned int  su4;
typedef __attribute__((ext_vector_type(8)))  unsigned int  su8;

constexpr int BATCH = 4, SEQ = 2048, HEADS = 16, DH = 64, DM = 1024;
constexpr int TOK = BATCH * SEQ;   // 8192 token rows

static __device__ __forceinline__ unsigned short f2bf(float f) {
  unsigned int u = __builtin_bit_cast(unsigned int, f);
  u += 0x7fffu + ((u >> 16) & 1u);          // round-to-nearest-even
  return (unsigned short)(u >> 16);
}

union AB16 { v16bf v; uint4 q[2]; };         // one 16x32 / 32x16 bf16 fragment

// Low 32 bits of a generic pointer to __shared__ == LDS byte offset
// (flat->LDS aperture).
static __device__ __forceinline__ unsigned lds_off(const void* p) {
  return (unsigned)(uintptr_t)p;
}

// GLOBAL_LOAD_ASYNC_TO_LDS_B128, GVS form: mem = saddr + voff; tracked ASYNCcnt.
static __device__ __forceinline__ void async_copy_b128(const void* gbase,
                                                       unsigned byte_off,
                                                       void* lds_dst) {
  unsigned d = lds_off(lds_dst);
  asm volatile("global_load_async_to_lds_b128 %0, %1, %2"
               :: "v"(d), "v"(byte_off), "s"(gbase)
               : "memory");
}
static __device__ __forceinline__ void wait_async0() {
  asm volatile("s_wait_asynccnt 0x0" ::: "memory");
}

// Tensor Data Mover: 2D tile (tile0 x tile1 elements, 2-byte data) -> LDS.
static __device__ __forceinline__ void tdm_load_tile_2d(unsigned lds_base,
                                                        const void* gaddr,
                                                        unsigned group1_w0,
                                                        unsigned dim0, unsigned dim1,
                                                        unsigned tile0, unsigned tile1,
                                                        unsigned stride0) {
  unsigned long long ga = (unsigned long long)(uintptr_t)gaddr;
  su4 g0;
  g0[0] = 1u;                                   // count=1, user-mode descriptor
  g0[1] = lds_base;                             // lds_addr [63:32]
  g0[2] = (unsigned)ga;                         // global_addr lo
  g0[3] = (unsigned)(ga >> 32) | 0x80000000u;   // global_addr hi | type=2<<30
  su8 g1;
  g1[0] = group1_w0;                            // mask=0|data_size|pad ctl
  g1[1] = (dim0 & 0xFFFFu) << 16;               // tensor_dim0 [79:48] lo
  g1[2] = (dim0 >> 16) | ((dim1 & 0xFFFFu) << 16); // dim0 hi | tensor_dim1 lo
  g1[3] = (dim1 >> 16) | (tile0 << 16);         // dim1 hi | tile_dim0
  g1[4] = tile1;                                // tile_dim1 | tile_dim2(0)<<16
  g1[5] = stride0;                              // tensor_dim0_stride lo32
  g1[6] = 0u;
  g1[7] = 0u;
  asm volatile("tensor_load_to_lds %0, %1" :: "s"(g0), "s"(g1) : "memory");
}

// TDM group1 word0: data_size=1 (2B), pad_enable, pad_interval=4 (32 dwords =
// one 128B row), pad_amount=3 (4 dwords = 16B)  ->  LDS row stride 144B.
constexpr unsigned TDM_G1W0_PAD =
    (1u << 16) | (1u << 20) | (4u << 22) | (3u << 25);

// ---------------------------------------------------------------------------
// Kernel 1: activations -> bf16   (qk = x + pos_embed, xv = x)
// ---------------------------------------------------------------------------
__global__ void prep_act(const float* __restrict__ x, const float* __restrict__ pe,
                         unsigned short* __restrict__ xb, unsigned short* __restrict__ qkb,
                         int n) {
  int i = blockIdx.x * blockDim.x + threadIdx.x;
  if (i < n) {
    float xv = x[i];
    xb[i]  = f2bf(xv);
    qkb[i] = f2bf(xv + pe[i]);
  }
}

// ---------------------------------------------------------------------------
// Kernel 2: weights -> n-major bf16 B-matrices [N=1024][K=1024]
//   W_{Q,K,V}[i,h,m] flat == [n=i*64+h][k=m]  -> straight cast.
//   W_O[i,m,h]: n = m, k = i*64+h             -> gather.
// ---------------------------------------------------------------------------
__global__ void prep_w(const float* __restrict__ Wq, const float* __restrict__ Wk,
                       const float* __restrict__ Wv, const float* __restrict__ Wo,
                       unsigned short* __restrict__ WqT, unsigned short* __restrict__ WkT,
                       unsigned short* __restrict__ WvT, unsigned short* __restrict__ WoT) {
  int i = blockIdx.x * blockDim.x + threadIdx.x;   // over 1024*1024
  int n = i >> 10, k = i & 1023;
  WqT[i] = f2bf(Wq[i]);
  WkT[i] = f2bf(Wk[i]);
  WvT[i] = f2bf(Wv[i]);
  WoT[i] = f2bf(Wo[(size_t)(k >> 6) * (1024 * 64) + (size_t)n * 64 + (k & 63)]);
}

// ---------------------------------------------------------------------------
// Kernel 3: GEMM  C[8192x1024] = A[8192x1024] * B^T (B stored [N][K], bf16)
//   Double-buffered LDS; both tiles staged with async global->LDS b128.
//   MODE 0: f32 row-major  |  MODE 1: bf16 [B,H,S,64]  |  MODE 2: bf16 [B,H,64,S]
// ---------------------------------------------------------------------------
template <int MODE>
__global__ __launch_bounds__(256)
void gemm_bf16(const unsigned short* __restrict__ A,
               const unsigned short* __restrict__ Bm,
               float* __restrict__ Cf, unsigned short* __restrict__ Cq,
               float scale) {
  __shared__ __align__(16) unsigned short As[2][128][40];   // [m][k], pad 32->40
  __shared__ __align__(16) unsigned short Bs[2][128][40];   // [n][k], pad 32->40

  const int tid  = threadIdx.x;
  const int lane = tid & 31;
  const int wid  = tid >> 5;
  const int wm   = wid >> 1;                 // 0..3 : 32-row strip
  const int wn   = wid & 1;                  // 0..1 : 64-col strip
  const int m0   = blockIdx.y * 128;
  const int n0   = blockIdx.x * 128;

  // Stage one 128x32 A tile + 128x32 B tile into LDS buffer `bufi` (async).
  auto stage = [&](int k0, int bufi) {
    int ra = tid >> 2;                 // 0..63
    int ca = (tid & 3) * 8;            // 0,8,16,24
    for (int p = 0; p < 2; ++p) {
      unsigned off = (unsigned)(((m0 + ra + p * 64) * DM + k0 + ca) * 2);
      async_copy_b128(A, off, &As[bufi][ra + p * 64][ca]);
    }
    int nb = tid >> 1;                 // 0..127
    int cb = (tid & 1) * 16;           // 0,16
    for (int p = 0; p < 2; ++p) {
      int c = cb + p * 8;
      unsigned off = (unsigned)(((n0 + nb) * DM + k0 + c) * 2);
      async_copy_b128(Bm, off, &Bs[bufi][nb][c]);
    }
  };

  v8f acc[2][4];
  for (int f = 0; f < 2; ++f)
    for (int g = 0; g < 4; ++g)
      for (int r = 0; r < 8; ++r) acc[f][g][r] = 0.f;

  stage(0, 0);
  for (int k0 = 0, it = 0; k0 < DM; k0 += 32, ++it) {
    wait_async0();
    __syncthreads();
    const int cur = it & 1;
    if (k0 + 32 < DM) stage(k0 + 32, cur ^ 1);   // overlap DMA with WMMA
    if (k0 + 64 < DM) {                          // prefetch tile after next
      __builtin_prefetch(A  + (size_t)(m0 + (tid >> 1)) * DM + k0 + 64, 0, 1);
      __builtin_prefetch(Bm + (size_t)(n0 + (tid >> 1)) * DM + k0 + 64, 0, 1);
    }

    AB16 af[2];
    for (int f = 0; f < 2; ++f) {
      int m  = wm * 32 + f * 16 + (lane & 15);
      int c0 = (lane >> 4) * 8;      // ISA A layout: K 0-7/16-23 vs 8-15/24-31
      af[f].q[0] = *reinterpret_cast<const uint4*>(&As[cur][m][c0]);
      af[f].q[1] = *reinterpret_cast<const uint4*>(&As[cur][m][c0 + 16]);
    }
    for (int g = 0; g < 4; ++g) {
      AB16 bfr;
      int n  = wn * 64 + g * 16 + (lane & 15);
      int kc = (lane >> 4) * 8;      // B layout matches A: K 0-7/16-23 vs 8-15/24-31
      bfr.q[0] = *reinterpret_cast<const uint4*>(&Bs[cur][n][kc]);
      bfr.q[1] = *reinterpret_cast<const uint4*>(&Bs[cur][n][kc + 16]);
      for (int f = 0; f < 2; ++f)
        acc[f][g] = __builtin_amdgcn_wmma_f32_16x16x32_bf16(
            false, af[f].v, false, bfr.v, (short)0, acc[f][g], false, false);
    }
  }

  // ---- store (C/D layout: VGPR r -> M = r + 8*(lane>=16), N = lane&15) ----
  for (int f = 0; f < 2; ++f)
    for (int g = 0; g < 4; ++g) {
      const int n  = n0 + wn * 64 + g * 16 + (lane & 15);
      const int mb = m0 + wm * 32 + f * 16 + ((lane >> 4) << 3);
      if constexpr (MODE == 0) {
        float* dst = Cf + (size_t)mb * DM + n;
        for (int r = 0; r < 8; ++r)
          dst[(size_t)r * DM] = acc[f][g][r] * scale;
      } else {
        const int b = mb >> 11, p = mb & (SEQ - 1);   // p advances by 1 per r
        const int head = n >> 6, h = n & 63;
        if constexpr (MODE == 1) {                    // [B,H,S,64]
          unsigned short* dst =
              Cq + ((size_t)(b * HEADS + head) * SEQ + p) * DH + h;
          for (int r = 0; r < 8; ++r)
            dst[(size_t)r * DH] = f2bf(acc[f][g][r] * scale);
        } else {                                      // MODE 2: [B,H,64,S]
          unsigned short* dst =
              Cq + ((size_t)((b * HEADS + head) * DH + h)) * SEQ + p;
          for (int r = 0; r < 8; ++r)
            dst[r] = f2bf(acc[f][g][r] * scale);
        }
      }
    }
}

// ---------------------------------------------------------------------------
// Kernel 4: flash attention, one WG per (b, head, 128-query tile)
//   K tile (wave 0) and pre-transposed V tile (wave 1) both staged by TDM
//   descriptors into double-buffered LDS; descriptor pad gives 72-short rows.
//   Compute skipped above the diagonal; mask applied only on diagonal tiles.
// ---------------------------------------------------------------------------
__global__ __launch_bounds__(256)
void flash_attn(const unsigned short* __restrict__ Qb,
                const unsigned short* __restrict__ Kb,
                const unsigned short* __restrict__ Vb,   // [B,H,64,SEQ]
                unsigned short* __restrict__ Zb) {
  __shared__ __align__(16) unsigned short Ks[2][64][72];   // [key][h]
  __shared__ __align__(16) unsigned short Vt[2][64][72];   // [h][key]
  __shared__ __align__(16) unsigned short Ps[8][16][72];   // per-wave P staging

  const int tid  = threadIdx.x;
  const int lane = tid & 31;
  const int wid  = tid >> 5;
  const int qt   = blockIdx.x;        // query tile (128 rows)
  const int head = blockIdx.y;
  const int b    = blockIdx.z;
  const size_t hb   = (size_t)(b * HEADS + head) * SEQ;    // K row base
  const int   qbase = qt * 128 + wid * 16;
  const unsigned ksBase = lds_off(&Ks[0][0][0]);
  const unsigned vtBase = lds_off(&Vt[0][0][0]);
  constexpr unsigned TILE_BYTES = 64 * 72 * 2;             // one LDS buffer
  const unsigned short* Vg = Vb + (size_t)(b * HEADS + head) * DH * SEQ;

  auto issue_tiles = [&](int kt, int bufi) {
    if (wid == 0) {        // K: 64 keys x 64 h, rows contiguous (stride 64)
      tdm_load_tile_2d(ksBase + bufi * TILE_BYTES,
                       Kb + (hb + (size_t)kt * 64) * DH, TDM_G1W0_PAD,
                       64, 64, 64, 64, /*stride0=*/64);
    } else if (wid == 1) { // V^T: 64 h x 64 keys, row stride SEQ
      tdm_load_tile_2d(vtBase + bufi * TILE_BYTES,
                       Vg + (size_t)kt * 64, TDM_G1W0_PAD,
                       64, 64, 64, 64, /*stride0=*/(unsigned)SEQ);
    }
  };

  // Q fragments for this wave's 16 rows (two K-steps over Dh=64)
  AB16 qf[2];
  {
    const unsigned short* qrow = Qb + (hb + qbase + (lane & 15)) * DH;
    int c0 = (lane >> 4) * 8;
    for (int ks = 0; ks < 2; ++ks) {
      qf[ks].q[0] = *reinterpret_cast<const uint4*>(qrow + ks * 32 + c0);
      qf[ks].q[1] = *reinterpret_cast<const uint4*>(qrow + ks * 32 + c0 + 16);
    }
  }

  float mrow[8], lrow[8];
  v8f oacc[4];
  for (int r = 0; r < 8; ++r) { mrow[r] = -3e38f; lrow[r] = 0.f; }
  for (int nt = 0; nt < 4; ++nt)
    for (int r = 0; r < 8; ++r) oacc[nt][r] = 0.f;

  const int qrow0 = qbase + ((lane >> 4) << 3);
  const int nkt   = 2 * qt + 2;                     // causal: keys <= last query

  issue_tiles(0, 0);
  for (int kt = 0; kt < nkt; ++kt) {
    __builtin_amdgcn_s_wait_tensorcnt(0);           // no-op for waves 2..7
    __syncthreads();
    const int cur = kt & 1;
    if (kt + 1 < nkt) issue_tiles(kt + 1, cur ^ 1); // overlap TDM with WMMA

    if (kt * 64 <= qbase + 15) {                    // tile reaches this wave
      const bool diag = (kt * 64 + 63 > qbase);     // tile crosses diagonal

      // ---- scores S = Q * K^T  (16x64 per wave) ----
      v8f sacc[4];
      for (int g = 0; g < 4; ++g)
        for (int r = 0; r < 8; ++r) sacc[g][r] = 0.f;
      for (int ks = 0; ks < 2; ++ks)
        for (int g = 0; g < 4; ++g) {
          AB16 bk;
          int key = g * 16 + (lane & 15);
          int hc  = ks * 32 + ((lane >> 4) << 3);
          bk.q[0] = *reinterpret_cast<const uint4*>(&Ks[cur][key][hc]);
          bk.q[1] = *reinterpret_cast<const uint4*>(&Ks[cur][key][hc + 16]);
          sacc[g] = __builtin_amdgcn_wmma_f32_16x16x32_bf16(
              false, qf[ks].v, false, bk.v, (short)0, sacc[g], false, false);
        }

      // ---- mask (diagonal tile only) + per-tile row max ----
      float tmax[8];
      if (diag) {
        for (int r = 0; r < 8; ++r) {
          int q = qrow0 + r;
          float mx = -3e38f;
          for (int g = 0; g < 4; ++g) {
            int key = kt * 64 + g * 16 + (lane & 15);
            float s = (key <= q) ? sacc[g][r] : -3e38f;
            sacc[g][r] = s;
            mx = fmaxf(mx, s);
          }
          tmax[r] = mx;
        }
      } else {
        for (int r = 0; r < 8; ++r)
          tmax[r] = fmaxf(fmaxf(sacc[0][r], sacc[1][r]),
                          fmaxf(sacc[2][r], sacc[3][r]));
      }
      for (int off = 1; off < 16; off <<= 1)        // reduce over 16-lane half
        for (int r = 0; r < 8; ++r)
          tmax[r] = fmaxf(tmax[r], __shfl_xor(tmax[r], off, 32));

      // ---- online softmax ----
      float alpha[8], rsum[8];
      for (int r = 0; r < 8; ++r) {
        float mn = fmaxf(mrow[r], tmax[r]);
        alpha[r] = __expf(mrow[r] - mn);
        mrow[r]  = mn;
        float sum = 0.f;
        for (int g = 0; g < 4; ++g) {
          float pe = __expf(sacc[g][r] - mn);       // masked -> exp(-huge) = 0
          sacc[g][r] = pe;
          sum += pe;
        }
        rsum[r] = sum;
      }
      for (int off = 1; off < 16; off <<= 1)
        for (int r = 0; r < 8; ++r)
          rsum[r] += __shfl_xor(rsum[r], off, 32);
      for (int r = 0; r < 8; ++r) {
        lrow[r] = lrow[r] * alpha[r] + rsum[r];
        for (int nt = 0; nt < 4; ++nt) oacc[nt][r] *= alpha[r];
      }

      // ---- C-layout P -> A-layout via per-wave LDS patch ----
      for (int g = 0; g < 4; ++g)
        for (int r = 0; r < 8; ++r)
          Ps[wid][((lane >> 4) << 3) + r][g * 16 + (lane & 15)] =
              f2bf(sacc[g][r]);

      // ---- O += P * V ----
      for (int ks = 0; ks < 2; ++ks) {
        AB16 ap;
        int m  = lane & 15;
        int c0 = ks * 32 + ((lane >> 4) << 3);
        ap.q[0] = *reinterpret_cast<const uint4*>(&Ps[wid][m][c0]);
        ap.q[1] = *reinterpret_cast<const uint4*>(&Ps[wid][m][c0 + 16]);
        for (int nt = 0; nt < 4; ++nt) {
          AB16 bv;
          int h  = nt * 16 + (lane & 15);
          int kc = ks * 32 + ((lane >> 4) << 3);
          bv.q[0] = *reinterpret_cast<const uint4*>(&Vt[cur][h][kc]);
          bv.q[1] = *reinterpret_cast<const uint4*>(&Vt[cur][h][kc + 16]);
          oacc[nt] = __builtin_amdgcn_wmma_f32_16x16x32_bf16(
              false, ap.v, false, bv.v, (short)0, oacc[nt], false, false);
        }
      }
    }
  }

  // ---- normalize and write z[b,q, head*64+h] as bf16 for the W_O GEMM ----
  unsigned short* zr = Zb + (size_t)(b * SEQ + qrow0) * DM + head * DH;
  for (int nt = 0; nt < 4; ++nt) {
    int h = nt * 16 + (lane & 15);
    for (int r = 0; r < 8; ++r)
      zr[(size_t)r * DM + h] = f2bf(oacc[nt][r] / lrow[r]);
  }
}

// ---------------------------------------------------------------------------
// Host-side launcher
// ---------------------------------------------------------------------------
extern "C" void kernel_launch(void* const* d_in, const int* in_sizes, int n_in,
                              void* d_out, int out_size, void* d_ws, size_t ws_size,
                              hipStream_t stream) {
  (void)in_sizes; (void)n_in; (void)out_size; (void)ws_size;
  const float* x  = (const float*)d_in[0];
  const float* pe = (const float*)d_in[1];
  const float* Wq = (const float*)d_in[2];
  const float* Wk = (const float*)d_in[3];
  const float* Wv = (const float*)d_in[4];
  const float* Wo = (const float*)d_in[5];

  char* ws = (char*)d_ws;
  const size_t MB = 1024 * 1024;
  unsigned short* qkb = (unsigned short*)(ws +  0 * MB);  // 16 MB
  unsigned short* xb  = (unsigned short*)(ws + 16 * MB);  // 16 MB
  unsigned short* WqT = (unsigned short*)(ws + 32 * MB);  //  2 MB [N][K]
  unsigned short* WkT = (unsigned short*)(ws + 34 * MB);  //  2 MB [N][K]
  unsigned short* WvT = (unsigned short*)(ws + 36 * MB);  //  2 MB [N][K]
  unsigned short* WoT = (unsigned short*)(ws + 38 * MB);  //  2 MB [N][K]
  unsigned short* Qb  = (unsigned short*)(ws + 40 * MB);  // 16 MB [B,H,S,64]
  unsigned short* Kb  = (unsigned short*)(ws + 56 * MB);  // 16 MB [B,H,S,64]
  unsigned short* Vb  = (unsigned short*)(ws + 72 * MB);  // 16 MB [B,H,64,S]
  unsigned short* Zb  = (unsigned short*)(ws + 88 * MB);  // 16 MB [TOK,1024]

  int nact = TOK * DM;
  prep_act<<<(nact + 255) / 256, 256, 0, stream>>>(x, pe, xb, qkb, nact);
  prep_w<<<(DM * DM) / 256, 256, 0, stream>>>(Wq, Wk, Wv, Wo, WqT, WkT, WvT, WoT);

  dim3 ggrid(DM / 128, TOK / 128);   // (8, 64)
  gemm_bf16<1><<<ggrid, 256, 0, stream>>>(qkb, WqT, nullptr, Qb, 0.125f); // Q
  gemm_bf16<1><<<ggrid, 256, 0, stream>>>(qkb, WkT, nullptr, Kb, 1.0f);   // K
  gemm_bf16<2><<<ggrid, 256, 0, stream>>>(xb,  WvT, nullptr, Vb, 1.0f);   // V^T

  flash_attn<<<dim3(SEQ / 128, HEADS, BATCH), 256, 0, stream>>>(Qb, Kb, Vb, Zb);

  gemm_bf16<0><<<ggrid, 256, 0, stream>>>(Zb, WoT, (float*)d_out, nullptr, 1.0f);
}
